// SequentialRNN_89206470738234
// MI455X (gfx1250) — matrix operand order repaired and checked by reading, not compiled, not run on hardware
//
#include <hip/hip_runtime.h>

typedef __bf16 bf16;
typedef bf16  v16bf __attribute__((ext_vector_type(16)));
typedef bf16  v8bf  __attribute__((ext_vector_type(8)));
typedef float v8f   __attribute__((ext_vector_type(8)));

namespace {
constexpr int B    = 64;
constexpr int S    = 2048;
constexpr int H    = 1024;
constexpr int NWG  = 32;    // persistent workgroups
constexpr int TPB  = 256;   // 8 waves (wave32) per workgroup
constexpr int COLS = 32;    // N-columns of W^T owned per workgroup
}

__device__ __forceinline__ float fast_tanh(float v) {
#if __has_builtin(__builtin_amdgcn_tanhf)
  return __builtin_amdgcn_tanhf(v);   // single V_TANH_F32 (CDNA5 TRANS op)
#else
  return tanhf(v);
#endif
}

__device__ __forceinline__ void grid_sync(unsigned* cnt, unsigned* gen) {
  __threadfence();          // make this step's h writes device-visible
  __syncthreads();
  if (threadIdx.x == 0) {
    unsigned g   = __hip_atomic_load(gen, __ATOMIC_ACQUIRE, __HIP_MEMORY_SCOPE_AGENT);
    unsigned old = __hip_atomic_fetch_add(cnt, 1u, __ATOMIC_ACQ_REL, __HIP_MEMORY_SCOPE_AGENT);
    if (old == (unsigned)(NWG - 1)) {
      __hip_atomic_store(cnt, 0u, __ATOMIC_RELAXED, __HIP_MEMORY_SCOPE_AGENT);
      __hip_atomic_fetch_add(gen, 1u, __ATOMIC_RELEASE, __HIP_MEMORY_SCOPE_AGENT);
    } else {
      while (__hip_atomic_load(gen, __ATOMIC_ACQUIRE, __HIP_MEMORY_SCOPE_AGENT) == g)
        __builtin_amdgcn_s_sleep(1);
    }
  }
  __syncthreads();
}

// W_hh -> bf16; x[B,S] -> xt[S,B]; h0 broadcast into bf16 state; reset barrier.
__global__ __launch_bounds__(TPB) void rnn_prep(
    const float* __restrict__ W, const float* __restrict__ h0,
    const float* __restrict__ x,
    bf16* __restrict__ Wb, bf16* __restrict__ hbf0, float* __restrict__ xt,
    unsigned* __restrict__ cnt, unsigned* __restrict__ gen) {
  int idx = blockIdx.x * TPB + threadIdx.x;
  if (idx < H * H) Wb[idx] = (bf16)W[idx];
  if (idx < B * S) {                        // idx = b*S + t  (coalesced read)
    int b = idx >> 11;                      // / S
    int t = idx & (S - 1);
    xt[t * B + b] = x[idx];
  }
  if (idx < B * H) hbf0[idx] = (bf16)h0[idx & (H - 1)];
  if (idx == 0) { *cnt = 0u; *gen = 0u; }
}

__global__ __launch_bounds__(TPB) void rnn_main(
    const float* __restrict__ xt, const float* __restrict__ Wih,
    const float* __restrict__ dtp, const float* __restrict__ h0,
    const bf16* __restrict__ Wb, bf16* __restrict__ hbf,
    float* __restrict__ out, unsigned* cnt, unsigned* gen) {
  __shared__ bf16 ldsW[COLS * H];   // 64 KB: this WG's 32-column slice of W^T

  const int g    = blockIdx.x;
  const int tid  = threadIdx.x;
  const int wave = tid >> 5;
  const int lane = tid & 31;
  const int n_base = g * COLS;

  // Stage W^T slice into LDS once: ldsW[c*H + k] = W_hh[n_base+c][k]
  for (int i = tid; i < COLS * H; i += TPB) {
    int c = i >> 10;          // i / H
    int k = i & (H - 1);      // i % H
    ldsW[i] = Wb[(n_base + c) * H + k];
  }
  __syncthreads();

  // Tile assignment: wave -> (m-tile, local n-tile); one 16x16 D tile per wave.
  const int m0   = (wave & 3) * 16;
  const int cloc = (wave >> 2) * 16 + (lane & 15);   // local column 0..31
  const int j    = n_base + cloc;                    // output column
  const int row  = m0 + (lane & 15);                 // A-matrix row (batch idx)
  const int aoff = (lane < 16) ? 0 : 8;              // 16-bit A layout K sub-offset
  const int kb2  = (lane < 16) ? 0 : 8;              // 16-bit B layout K sub-offset
  const bool hi  = lane >= 16;
  const int mrow = m0 + (hi ? 8 : 0);                // first batch row of my D half

  const float dtv  = dtp[0];
  const float ts   = 0.001f + 0.099f * (1.0f / (1.0f + __expf(-dtv)));
  const float wihj = Wih[j];

  // f32 state lives in registers: this lane owns h[mrow..mrow+7][j] forever.
  float hreg[8];
  {
    const float h0j = h0[j];
    #pragma unroll
    for (int i = 0; i < 8; ++i) hreg[i] = h0j;
  }

  const bf16* ldsB = &ldsW[cloc * H];

  for (int t = 0; t < S; ++t) {
    const int cur = t & 1, nxt = cur ^ 1;
    const bf16* hbf_r = hbf + cur * (B * H);
    bf16*       hbf_w = hbf + nxt * (B * H);

    v8f acc = {0.f, 0.f, 0.f, 0.f, 0.f, 0.f, 0.f, 0.f};
    const bf16* arow = hbf_r + row * H + aoff;

    #pragma unroll 4
    for (int kk = 0; kk < 32; ++kk) {
      const int k0 = kk * 32;
      // A fragment (16x32 bf16): per lane, K = {aoff..+7} and {aoff+16..+23}
      union { v16bf v; v8bf h[2]; } a;
      a.h[0] = *(const v8bf*)(arow + k0);
      a.h[1] = *(const v8bf*)(arow + k0 + 16);
      // B fragment (32x16 bf16) from LDS: per lane, K = {kb2..+7} and {kb2+16..+23}
      union { v16bf v; v8bf h[2]; } bu;
      bu.h[0] = *(const v8bf*)(ldsB + k0 + kb2);
      bu.h[1] = *(const v8bf*)(ldsB + k0 + kb2 + 16);
      v16bf b = bu.v;
      acc = __builtin_amdgcn_wmma_f32_16x16x32_bf16(false, a.v, false, b,
                                                    (short)0, acc, false, false);
    }

    // x_t for my 8 batch rows: one contiguous 32B load (xt is [S,B])
    const v8f xv8 = *(const v8f*)(xt + t * B + mrow);

    // Epilogue: h_new = h + tanh(ts*(hW^T + W_ih*x_t)), all in registers
    #pragma unroll
    for (int i = 0; i < 8; ++i) {
      const int b_row = mrow + i;
      const float pre = ts * (acc[i] + wihj * xv8[i]);
      const float hn  = hreg[i] + fast_tanh(pre);
      hreg[i] = hn;
      hbf_w[b_row * H + j] = (bf16)hn;                            // feed next step's A
      __builtin_nontemporal_store(hn, &out[((size_t)b_row * S + t) * H + j]);
    }

    grid_sync(cnt, gen);   // release h_{t+1} to all workgroups
  }
}

extern "C" void kernel_launch(void* const* d_in, const int* in_sizes, int n_in,
                              void* d_out, int out_size, void* d_ws, size_t ws_size,
                              hipStream_t stream) {
  const float* x   = (const float*)d_in[0];   // [B,S,1]
  const float* Whh = (const float*)d_in[1];   // [H,H]
  const float* Wih = (const float*)d_in[2];   // [1,H]
  const float* dt  = (const float*)d_in[3];   // scalar
  const float* h0  = (const float*)d_in[4];   // [H]
  float* out = (float*)d_out;

  char* ws = (char*)d_ws;
  constexpr size_t WB_BYTES  = (size_t)H * H * sizeof(bf16);   // 2 MB
  constexpr size_t HBF_BYTES = 2ull * B * H * sizeof(bf16);    // 256 KB
  constexpr size_t XT_BYTES  = (size_t)B * S * sizeof(float);  // 512 KB
  bf16*     Wb  = (bf16*)ws;
  bf16*     hbf = (bf16*)(ws + WB_BYTES);
  float*    xt  = (float*)(ws + WB_BYTES + HBF_BYTES);
  unsigned* cnt = (unsigned*)(ws + WB_BYTES + HBF_BYTES + XT_BYTES);
  unsigned* gen = cnt + 1;

  (void)in_sizes; (void)n_in; (void)out_size; (void)ws_size;

  rnn_prep<<<(H * H + TPB - 1) / TPB, TPB, 0, stream>>>(Whh, h0, x, Wb, hbf, xt, cnt, gen);
  rnn_main<<<NWG, TPB, 0, stream>>>(xt, Wih, dt, h0, Wb, hbf, out, cnt, gen);
}
